// Qwen3VLWithExpertModel_5171140624949
// MI455X (gfx1250) — matrix operand order repaired and checked by reading, not compiled, not run on hardware
//
#include <hip/hip_runtime.h>
#include <hip/hip_bf16.h>
#include <cstdint>
#include <cstddef>

// ---------------------------------------------------------------------------
// Types for CDNA5 WMMA (wave32, v_wmma_f32_16x16x32_bf16)
// ---------------------------------------------------------------------------
typedef __bf16 bf16_t;
typedef __attribute__((ext_vector_type(16))) __bf16 v16bf;
typedef __attribute__((ext_vector_type(8)))  __bf16 v8bf;
typedef __attribute__((ext_vector_type(8)))  float  v8f;

union FragAB {
  v16bf    v;
  v8bf     h[2];
  unsigned u[8];
};

// ---------------------------------------------------------------------------
// Problem constants (reference shapes)
// ---------------------------------------------------------------------------
#define NB        2        // batch
#define ST        1344     // total sequence (1024+256+64)
#define NH        16       // query heads
#define NKV       8        // kv heads
#define HD        128      // head dim
#define ATT_SCALE 0.08838834764831845f   // 128^-0.5
#define RMS_EPS   1e-6f
#define ROPE_LOG_THETA 13.815510557964274f  // ln(1e6)

// ---------------------------------------------------------------------------
// fp32 -> bf16 conversion (one-time weight repack; halves HBM bytes,
// makes every GEMM a v_wmma_f32_16x16x32_bf16 stream)
// ---------------------------------------------------------------------------
__global__ void cvt_bf16_kernel(const float* __restrict__ s,
                                bf16_t* __restrict__ d, long n) {
  long i = (long)blockIdx.x * 256 + threadIdx.x;
  if (i < n) d[i] = (bf16_t)s[i];
}

// ---------------------------------------------------------------------------
// Row RMSNorm: out_bf (bf16, GEMM input) or out_f (fp32, final output)
// ---------------------------------------------------------------------------
__global__ __launch_bounds__(256) void rmsnorm_rows_kernel(
    const float* __restrict__ x, const float* __restrict__ w,
    bf16_t* __restrict__ out_bf, float* __restrict__ out_f, int hdim) {
  long row = blockIdx.x;
  const float* xr = x + row * (long)hdim;
  float ss = 0.f;
  for (int i = threadIdx.x; i < hdim; i += 256) { float v = xr[i]; ss += v * v; }
  __shared__ float sred[8];
  #pragma unroll
  for (int off = 1; off < 32; off <<= 1) ss += __shfl_xor(ss, off, 32);
  if ((threadIdx.x & 31) == 0) sred[threadIdx.x >> 5] = ss;
  __syncthreads();
  float tot = 0.f;
  #pragma unroll
  for (int i = 0; i < 8; ++i) tot += sred[i];
  float inv = rsqrtf(tot / (float)hdim + RMS_EPS);
  for (int i = threadIdx.x; i < hdim; i += 256) {
    float v = xr[i] * inv * w[i];
    if (out_bf) out_bf[row * (long)hdim + i] = (bf16_t)v;
    else        out_f [row * (long)hdim + i] = v;
  }
}

// ---------------------------------------------------------------------------
// NT GEMM: C[M,N] = A[M,K] * B[N,K]^T (+Res), A/B bf16, C fp32.
// Block tile 128x128 (256 thr = 8 waves, 4x2 wave grid, 32x64 wave tile),
// k-step 64 (two WMMA k-slices per LDS stage): 16 WMMAs per stage round,
// fragment reads are ds_load_b128 per the 16-bit A/B VGPR layouts.
// A-row remap: arow = (m / rpb) * bstride + roff + (m % rpb)
// ---------------------------------------------------------------------------
__global__ __launch_bounds__(256) void gemm_bf16_wmma_kernel(
    const bf16_t* __restrict__ A, const bf16_t* __restrict__ Bw,
    float* __restrict__ C, const float* __restrict__ Res,
    int M, int N, int K, int rpb, int bstride, int roff) {
  __shared__ bf16_t lA[128][64];
  __shared__ bf16_t lB[128][64];

  const int tid  = threadIdx.x;
  const int lane = tid & 31;
  const int wv   = tid >> 5;
  const int wr   = wv & 3;    // wave row (M): 4 x 32 rows
  const int wc   = wv >> 2;   // wave col (N): 2 x 64 cols
  const int m0   = blockIdx.x * 128;
  const int n0   = blockIdx.y * 128;
  const int hl   = lane >> 4;       // half of wave
  const int l15  = lane & 15;
  const int abase = hl * 8;         // A-frag K base per lane-half
  const int bbase = hl * 8;         // B-frag K base per lane-half

  v8f acc[2][4];
  #pragma unroll
  for (int i = 0; i < 2; ++i)
    #pragma unroll
    for (int j = 0; j < 4; ++j) acc[i][j] = (v8f){};

  for (int k0 = 0; k0 < K; k0 += 64) {
    // cooperative LDS fill: A/B each 128x64 bf16 = 1024 x b128, 4 per thread
    #pragma unroll
    for (int it = 0; it < 4; ++it) {
      int i  = tid + it * 256;
      int r  = i >> 3, kq = (i & 7) * 8;
      int ml = m0 + r;
      long arow = (long)(ml / rpb) * bstride + roff + (ml % rpb);
      *(v8bf*)(&lA[r][kq]) = *(const v8bf*)(A + arow * (long)K + k0 + kq);
      *(v8bf*)(&lB[r][kq]) = *(const v8bf*)(Bw + (long)(n0 + r) * K + k0 + kq);
    }
    // prefetch next k-tile into GL2 (global_prefetch_b8)
    if (k0 + 64 < K) {
      int r = tid >> 1, kq = (tid & 1) * 32;
      int ml = m0 + r;
      long arow = (long)(ml / rpb) * bstride + roff + (ml % rpb);
      __builtin_prefetch(A  + arow * (long)K + k0 + 64 + kq, 0, 0);
      __builtin_prefetch(Bw + (long)(n0 + r) * K + k0 + 64 + kq, 0, 0);
    }
    __syncthreads();

    #pragma unroll
    for (int ks = 0; ks < 2; ++ks) {
      FragAB af[2];
      #pragma unroll
      for (int i = 0; i < 2; ++i) {
        int am = wr * 32 + i * 16 + l15;
        af[i].h[0] = *(const v8bf*)(&lA[am][ks * 32 + abase]);
        af[i].h[1] = *(const v8bf*)(&lA[am][ks * 32 + abase + 16]);
      }
      #pragma unroll
      for (int j = 0; j < 4; ++j) {
        FragAB bf;
        int bn = wc * 64 + j * 16 + l15;
        bf.h[0] = *(const v8bf*)(&lB[bn][ks * 32 + bbase]);
        bf.h[1] = *(const v8bf*)(&lB[bn][ks * 32 + bbase + 16]);
        #pragma unroll
        for (int i = 0; i < 2; ++i)
          acc[i][j] = __builtin_amdgcn_wmma_f32_16x16x32_bf16(
              false, af[i].v, false, bf.v, (short)0, acc[i][j], false, false);
      }
    }
    __syncthreads();
  }

  // epilogue: C layout VGPR r -> row r + 8*half, col = lane&15
  #pragma unroll
  for (int i = 0; i < 2; ++i)
    #pragma unroll
    for (int j = 0; j < 4; ++j)
      #pragma unroll
      for (int r = 0; r < 8; ++r) {
        int ml  = m0 + wr * 32 + i * 16 + r + 8 * hl;
        int col = n0 + wc * 64 + j * 16 + l15;
        float v = acc[i][j][r];
        if (Res) v += Res[(long)ml * N + col];
        C[(long)ml * N + col] = v;
      }
}

// ---------------------------------------------------------------------------
// Q/K head-RMSNorm + RoPE + layout transform to [B, nH, ST, HD] bf16.
// nw != null -> rmsnorm(nw) + rope ; nw == null -> plain copy (V).
// src: [B*Sb, nH*HD] fp32 (QKV GEMM output)
// ---------------------------------------------------------------------------
__global__ __launch_bounds__(128) void qkv_post_kernel(
    const float* __restrict__ src, const float* __restrict__ nw,
    const int* __restrict__ pos, bf16_t* __restrict__ dst,
    int Sb, int soff, int nHh) {
  int row = blockIdx.x, hh = blockIdx.y, d = threadIdx.x;
  int b = row / Sb, sl = row % Sb, sg = soff + sl;
  float x = src[(long)row * (nHh * HD) + hh * HD + d];
  float out;
  __shared__ float sh[HD];
  if (nw) {
    sh[d] = x * x;
    __syncthreads();
    #pragma unroll
    for (int s2 = 64; s2 > 0; s2 >>= 1) {
      if (d < s2) sh[d] += sh[d + s2];
      __syncthreads();
    }
    float inv = rsqrtf(sh[0] / (float)HD + RMS_EPS);
    __syncthreads();
    float xn = x * inv * nw[d];
    sh[d] = xn;
    __syncthreads();
    int p = pos[b * ST + sg];
    int j = d & 63;
    float fr = (float)p * __expf(-((float)(2 * j) / (float)HD) * ROPE_LOG_THETA);
    float c = cosf(fr), s = sinf(fr);
    float other = (d < 64) ? -sh[d + 64] : sh[d - 64];
    out = xn * c + other * s;
  } else {
    out = x;
  }
  dst[(((long)b * nHh + hh) * ST + sg) * HD + d] = (bf16_t)out;
}

// ---------------------------------------------------------------------------
// Flash attention: 128 threads = 4 waves, each wave owns one 16-row q block
// (84 = 4*21 tiles -> even grid). Per wave: QK^T via 4 WMMAs (D=128),
// online softmax with wave32 shfl_xor row reductions (C row = r + 8*half),
// P staged in a per-wave LDS tile (K padded 16->32 with zeros), V staged
// transposed in per-wave LDS, PV via 8 WMMAs. GQA: kv head = h/2.
// Output att bf16 in [B, ST, NH*HD] (the O-projection input layout).
// ---------------------------------------------------------------------------
__global__ __launch_bounds__(128) void attn_wmma_kernel(
    const bf16_t* __restrict__ Q, const bf16_t* __restrict__ Kc,
    const bf16_t* __restrict__ Vc, const float* __restrict__ mask,
    bf16_t* __restrict__ att) {
  __shared__ bf16_t lp[4][16][32];
  __shared__ bf16_t lvt[4][HD][16];

  const int lane = threadIdx.x & 31;
  const int wv   = threadIdx.x >> 5;
  const int qt = blockIdx.x * 4 + wv, hq = blockIdx.y, b = blockIdx.z;
  const int kvh = hq >> 1;                 // NH/NKV == 2
  const int hl = lane >> 4, l15 = lane & 15;
  const int abase = hl * 8, bbase = hl * 8;

  // zero P tile once (cols 16..31 stay zero forever)
  for (int i = lane; i < 16 * 32; i += 32) ((bf16_t*)lp[wv])[i] = (bf16_t)0.0f;

  // Q fragments: 4 chunks of K=32 over D=128
  FragAB qf[4];
  const bf16_t* qbase = Q + (((long)(b * NH + hq)) * ST + qt * 16 + l15) * HD;
  #pragma unroll
  for (int c = 0; c < 4; ++c) {
    qf[c].h[0] = *(const v8bf*)(qbase + c * 32 + abase);
    qf[c].h[1] = *(const v8bf*)(qbase + c * 32 + abase + 16);
  }

  v8f acc[8];
  #pragma unroll
  for (int c = 0; c < 8; ++c) acc[c] = (v8f){};
  float mrow[8], lrow[8];
  #pragma unroll
  for (int r = 0; r < 8; ++r) { mrow[r] = -3.0e38f; lrow[r] = 0.f; }

  const bf16_t* kb = Kc + ((long)(b * NKV + kvh)) * ST * HD;
  const bf16_t* vb = Vc + ((long)(b * NKV + kvh)) * ST * HD;

  for (int kt = 0; kt < ST / 16; ++kt) {
    __syncthreads();   // previous iteration done with lvt/lp
    // stage V tile transposed: lvt[wv][d][key]
    #pragma unroll
    for (int t = 0; t < 8; ++t) {
      int vec = t * 32 + lane;             // 0..255
      int key = vec >> 4;
      int d0  = (vec & 15) * 8;
      v8bf vv = *(const v8bf*)(vb + (long)(kt * 16 + key) * HD + d0);
      #pragma unroll
      for (int e = 0; e < 8; ++e) lvt[wv][d0 + e][key] = vv[e];
    }
    // scores = Q K^T
    v8f sc = (v8f){};
    const bf16_t* krow = kb + (long)(kt * 16 + l15) * HD;
    #pragma unroll
    for (int c = 0; c < 4; ++c) {
      FragAB bf;
      bf.h[0] = *(const v8bf*)(krow + c * 32 + bbase);
      bf.h[1] = *(const v8bf*)(krow + c * 32 + bbase + 16);
      sc = __builtin_amdgcn_wmma_f32_16x16x32_bf16(
          false, qf[c].v, false, bf.v, (short)0, sc, false, false);
    }
    // online softmax; C row = r + 8*half, col = lane&15
    float sO[8];
    #pragma unroll
    for (int r = 0; r < 8; ++r) {
      int qrow = qt * 16 + r + 8 * hl;
      float v = sc[r] * ATT_SCALE +
                mask[((long)b * ST + qrow) * ST + kt * 16 + l15];
      float mx = v;
      #pragma unroll
      for (int off = 1; off < 16; off <<= 1) mx = fmaxf(mx, __shfl_xor(mx, off, 32));
      float mnew = fmaxf(mrow[r], mx);
      float p  = __expf(v - mnew);
      float ps = p;
      #pragma unroll
      for (int off = 1; off < 16; off <<= 1) ps += __shfl_xor(ps, off, 32);
      float so = __expf(mrow[r] - mnew);
      lrow[r] = lrow[r] * so + ps;
      mrow[r] = mnew;
      sO[r] = so;
      lp[wv][r + 8 * hl][l15] = (bf16_t)p;
    }
    __syncthreads();   // lp + lvt visible
    #pragma unroll
    for (int c = 0; c < 8; ++c)
      #pragma unroll
      for (int r = 0; r < 8; ++r) acc[c][r] *= sO[r];
    // PV
    FragAB pf;
    pf.h[0] = *(const v8bf*)(&lp[wv][l15][hl * 16]);
    pf.h[1] = *(const v8bf*)(&lp[wv][l15][hl * 16 + 8]);
    #pragma unroll
    for (int c = 0; c < 8; ++c) {
      FragAB vf;
      if (hl) {               // K rows 16..31 multiply zero P cols
        #pragma unroll
        for (int e = 0; e < 8; ++e) vf.u[e] = 0u;
      } else {
        vf.h[0] = *(const v8bf*)(&lvt[wv][c * 16 + l15][0]);
        vf.h[1] = *(const v8bf*)(&lvt[wv][c * 16 + l15][8]);
      }
      acc[c] = __builtin_amdgcn_wmma_f32_16x16x32_bf16(
          false, pf.v, false, vf.v, (short)0, acc[c], false, false);
    }
  }
  // write out: att[b, s, h*HD + d]
  #pragma unroll
  for (int c = 0; c < 8; ++c)
    #pragma unroll
    for (int r = 0; r < 8; ++r) {
      int s = qt * 16 + r + 8 * hl;
      float o = acc[c][r] / lrow[r];
      att[((long)b * ST + s) * (NH * HD) + hq * HD + c * 16 + l15] = (bf16_t)o;
    }
}

// ---------------------------------------------------------------------------
// SwiGLU activation: out_bf16 = silu(gate) * up
// ---------------------------------------------------------------------------
__global__ void silu_mul_kernel(const float* __restrict__ g,
                                const float* __restrict__ u,
                                bf16_t* __restrict__ o, long n) {
  long i = (long)blockIdx.x * 256 + threadIdx.x;
  if (i < n) {
    float gv = g[i];
    o[i] = (bf16_t)((gv / (1.f + __expf(-gv))) * u[i]);
  }
}

// ---------------------------------------------------------------------------
// Host orchestration
// ---------------------------------------------------------------------------
extern "C" void kernel_launch(void* const* d_in, const int* in_sizes, int n_in,
                              void* d_out, int out_size, void* d_ws, size_t ws_size,
                              hipStream_t stream) {
  (void)in_sizes; (void)n_in; (void)out_size; (void)ws_size;

  const float* x_b[3] = {(const float*)d_in[0], (const float*)d_in[1],
                         (const float*)d_in[2]};
  // expert params flattened depth-first: ln1,q,k,v,qn,kn,o,ln2,gate,up,down,norm
  const float* prm[3][12];
  for (int e = 0; e < 3; ++e)
    for (int j = 0; j < 12; ++j) prm[e][j] = (const float*)d_in[3 + e * 12 + j];
  const float* mask = (const float*)d_in[39];
  const int*   pos  = (const int*)d_in[40];

  const int  hs[3]   = {2048, 1024, 1024};
  const int  is_[3]  = {6144, 4096, 4096};
  const int  Sb[3]   = {1024, 256, 64};
  const int  soff[3] = {0, 1024, 1280};
  const long ooff[3] = {0, 2L * 1024 * 2048, 2L * 1024 * 2048 + 2L * 256 * 1024};

  char* wp = (char*)d_ws;
  auto alloc = [&](size_t bytes) -> void* {
    void* p = (void*)wp;
    wp += (bytes + 255) & ~(size_t)255;
    return p;
  };
  auto cvt = [&](const float* s, bf16_t* d, long n) {
    cvt_bf16_kernel<<<(unsigned)((n + 255) / 256), 256, 0, stream>>>(s, d, n);
  };
  auto gemm = [&](const bf16_t* A, const bf16_t* Bw, float* C, const float* Res,
                  int M, int N, int K, int rpb, int bstride, int roff) {
    gemm_bf16_wmma_kernel<<<dim3(M / 128, N / 128), 256, 0, stream>>>(
        A, Bw, C, Res, M, N, K, rpb, bstride, roff);
  };

  // --- bf16 weight repack (streams at half fp32 bytes; mostly L2-resident) ---
  bf16_t* W[3][7];  // q,k,v,o,gate,up,down
  const int widx[7] = {1, 2, 3, 6, 8, 9, 10};
  for (int e = 0; e < 3; ++e) {
    long sz[7] = {(long)NH * HD * hs[e],  (long)NKV * HD * hs[e],
                  (long)NKV * HD * hs[e], (long)hs[e] * NH * HD,
                  (long)is_[e] * hs[e],   (long)is_[e] * hs[e],
                  (long)hs[e] * is_[e]};
    for (int j = 0; j < 7; ++j) {
      W[e][j] = (bf16_t*)alloc((size_t)sz[j] * 2);
      cvt(prm[e][widx[j]], W[e][j], sz[j]);
    }
  }

  // --- activation workspace ---
  bf16_t* qb    = (bf16_t*)alloc((size_t)NB * NH  * ST * HD * 2);
  bf16_t* kbuf  = (bf16_t*)alloc((size_t)NB * NKV * ST * HD * 2);
  bf16_t* vbuf  = (bf16_t*)alloc((size_t)NB * NKV * ST * HD * 2);
  bf16_t* attb  = (bf16_t*)alloc((size_t)NB * ST * NH * HD * 2);
  bf16_t* hbuf  = (bf16_t*)alloc((size_t)2048 * 2048 * 2);
  bf16_t* mlpin = (bf16_t*)alloc((size_t)2048 * 6144 * 2);
  float*  qf    = (float*)alloc((size_t)2048 * 2048 * 4);
  float*  kf    = (float*)alloc((size_t)2048 * 1024 * 4);
  float*  vf    = (float*)alloc((size_t)2048 * 1024 * 4);
  float*  oout  = (float*)alloc((size_t)2048 * 2048 * 4);
  float*  gatef = (float*)alloc((size_t)2048 * 6144 * 4);
  float*  upf   = (float*)alloc((size_t)2048 * 6144 * 4);
  float*  xnew  = (float*)alloc((size_t)2048 * 2048 * 4);

  // --- per-expert: ln1 -> QKV GEMMs -> qk-norm + RoPE + layout ---
  for (int e = 0; e < 3; ++e) {
    int M = NB * Sb[e], hh = hs[e];
    rmsnorm_rows_kernel<<<M, 256, 0, stream>>>(x_b[e], prm[e][0], hbuf, nullptr, hh);
    gemm(hbuf, W[e][0], qf, nullptr, M, NH  * HD, hh, M, 0, 0);
    gemm(hbuf, W[e][1], kf, nullptr, M, NKV * HD, hh, M, 0, 0);
    gemm(hbuf, W[e][2], vf, nullptr, M, NKV * HD, hh, M, 0, 0);
    qkv_post_kernel<<<dim3(M, NH),  128, 0, stream>>>(qf, prm[e][4], pos, qb,   Sb[e], soff[e], NH);
    qkv_post_kernel<<<dim3(M, NKV), 128, 0, stream>>>(kf, prm[e][5], pos, kbuf, Sb[e], soff[e], NKV);
    qkv_post_kernel<<<dim3(M, NKV), 128, 0, stream>>>(vf, nullptr,   pos, vbuf, Sb[e], soff[e], NKV);
  }

  // --- shared attention over the full S=1344 sequence (84 = 4*21 q tiles) ---
  attn_wmma_kernel<<<dim3(ST / 64, NH, NB), 128, 0, stream>>>(qb, kbuf, vbuf, mask, attb);

  // --- per-expert: O proj (+residual) -> ln2 -> SwiGLU MLP (+residual) -> final norm
  for (int e = 0; e < 3; ++e) {
    int M = NB * Sb[e], hh = hs[e], II = is_[e];
    // O projection reads its slice of attb via the row remap
    gemm(attb, W[e][3], oout, x_b[e], M, hh, NH * HD, Sb[e], ST, soff[e]);
    rmsnorm_rows_kernel<<<M, 256, 0, stream>>>(oout, prm[e][7], hbuf, nullptr, hh);
    gemm(hbuf, W[e][4], gatef, nullptr, M, II, hh, M, 0, 0);
    gemm(hbuf, W[e][5], upf,   nullptr, M, II, hh, M, 0, 0);
    long n = (long)M * II;
    silu_mul_kernel<<<(unsigned)((n + 255) / 256), 256, 0, stream>>>(gatef, upf, mlpin, n);
    gemm(mlpin, W[e][6], xnew, oout, M, hh, II, M, 0, 0);
    rmsnorm_rows_kernel<<<M, 256, 0, stream>>>(xnew, prm[e][11], nullptr,
                                               (float*)d_out + ooff[e], hh);
  }
}
